// Autoformer_21320217657884
// MI455X (gfx1250) — compile-verified, matrix operand order checked by reading, not run on hardware
//
#include <hip/hip_runtime.h>
#include <math.h>

// ---------------------------------------------------------------------------
// Autoformer forward for MI455X (gfx1250, wave32, WMMA).
// All large matmuls run through v_wmma_f32_16x16x32_f16 (f16 in / f32 acc).
// Autocorrelation is computed as a batched Q*K^T WMMA GEMM + wrapped-diagonal
// reduction instead of an FFT (mathematically identical for the mean corr).
// GEMM: 128x64 block tile, 32x32 per-wave tile (4 accumulators, 8 WMMA/K-tile).
// ---------------------------------------------------------------------------

typedef __attribute__((ext_vector_type(16))) _Float16 v16h;
typedef __attribute__((ext_vector_type(8)))  float    v8f;

#define AF_B    32
#define AF_L    1024
#define AF_D    512
#define AF_FF   2048
#define AF_CIN  7
#define AF_MARK 4
#define AF_TOPK 6
#define AF_MA   25

#define BM 128
#define BN 64
#define BK 64
#define LDSS 72  // padded row stride (halves): 144B rows, 16B-aligned offsets

#define WMMA_F16(a, b, c) __builtin_amdgcn_wmma_f32_16x16x32_f16( \
    false, (a), false, (b), (short)0, (c), false, false)

// ------------------------------ WMMA GEMM ----------------------------------
// C[M,N] = A[M,K] * B[K,N] (or B[N,K] if TRANSB) with optional bias / exact
// GELU / residual. Row-major fp32 in/out, f16 WMMA math. Batched via gridDim.z.
// REQUIRES: M % 128 == 0, N % 64 == 0, K % 64 == 0 (true for all call sites).
template<bool TRANSB, bool BIAS, bool GELU_ACT, bool RES>
__global__ __launch_bounds__(256)
void gemm_wmma_f16(const float* __restrict__ A, const float* __restrict__ Bm,
                   const float* __restrict__ bias, const float* __restrict__ Res,
                   float* __restrict__ C, int M, int N, int Kd,
                   long long strideA, long long strideB, long long strideC)
{
    __shared__ __align__(16) _Float16 As[BM * LDSS];   // 18 KB
    __shared__ __align__(16) _Float16 Bs[BN * LDSS];   //  9 KB (transposed: Bs[n][k])

    const int tid    = threadIdx.x;
    const int wave   = tid >> 5;
    const int lane   = tid & 31;
    const int wm     = wave >> 1;        // 0..3  (32-row slab)
    const int wn     = wave & 1;         // 0..1  (32-col slab)
    const int lr     = lane & 15;
    const int half16 = lane >> 4;        // 0 or 1
    const int kb     = half16 << 3;      // 0 or 8 (ISA 16-bit fragment layout)
    const int z      = blockIdx.z;

    const float* Ab = A + (size_t)z * (size_t)strideA;
    const float* Bb = Bm + (size_t)z * (size_t)strideB;
    float*       Cb = C + (size_t)z * (size_t)strideC;
    const float* Rb = RES ? (Res + (size_t)z * (size_t)strideC) : nullptr;

    const int rowBase = blockIdx.y * BM;
    const int colBase = blockIdx.x * BN;

    v8f acc00 = {}, acc01 = {}, acc10 = {}, acc11 = {};

    for (int k0 = 0; k0 < Kd; k0 += BK) {
        // ---- stage 1: wide global loads into registers (b128, no LDS yet) ----
        float4 av[8], bv[4];
        #pragma unroll
        for (int it = 0; it < 8; ++it) {               // A: 128x64 fp32 = 2048 float4
            int idx = tid + it * 256;
            int r = idx >> 4, c4 = idx & 15;
            av[it] = *(const float4*)&Ab[(size_t)(rowBase + r) * Kd + (k0 + c4 * 4)];
        }
        #pragma unroll
        for (int it = 0; it < 4; ++it) {               // B: 64x64 fp32 = 1024 float4
            int idx = tid + it * 256;
            if (TRANSB) {
                int n = idx >> 4, c4 = idx & 15;
                bv[it] = *(const float4*)&Bb[(size_t)(colBase + n) * Kd + (k0 + c4 * 4)];
            } else {
                int kk = idx >> 4, n4 = idx & 15;
                bv[it] = *(const float4*)&Bb[(size_t)(k0 + kk) * N + (colBase + n4 * 4)];
            }
        }
        // prefetch next K-tile into near caches (gfx1250 global_prefetch_b8)
        if (k0 + BK < Kd) {
            __builtin_prefetch(&Ab[(size_t)(rowBase + (tid >> 1)) * Kd + (k0 + BK)], 0, 3);
            __builtin_prefetch(TRANSB
                ? &Bb[(size_t)(colBase + (tid >> 2)) * Kd + (k0 + BK)]
                : &Bb[(size_t)(k0 + BK + (tid >> 2)) * N + colBase], 0, 3);
        }

        __syncthreads();   // all waves finished reading previous LDS tiles

        // ---- stage 2: convert fp32 -> f16, store wide to LDS ----
        #pragma unroll
        for (int it = 0; it < 8; ++it) {
            int idx = tid + it * 256;
            int r = idx >> 4, c4 = idx & 15;
            union { _Float16 h[4]; uint2 u; } cv;
            cv.h[0] = (_Float16)av[it].x; cv.h[1] = (_Float16)av[it].y;
            cv.h[2] = (_Float16)av[it].z; cv.h[3] = (_Float16)av[it].w;
            *(uint2*)&As[r * LDSS + c4 * 4] = cv.u;
        }
        #pragma unroll
        for (int it = 0; it < 4; ++it) {
            int idx = tid + it * 256;
            if (TRANSB) {
                int n = idx >> 4, c4 = idx & 15;
                union { _Float16 h[4]; uint2 u; } cv;
                cv.h[0] = (_Float16)bv[it].x; cv.h[1] = (_Float16)bv[it].y;
                cv.h[2] = (_Float16)bv[it].z; cv.h[3] = (_Float16)bv[it].w;
                *(uint2*)&Bs[n * LDSS + c4 * 4] = cv.u;
            } else {
                int kk = idx >> 4, n4 = idx & 15;
                Bs[(n4 * 4 + 0) * LDSS + kk] = (_Float16)bv[it].x;
                Bs[(n4 * 4 + 1) * LDSS + kk] = (_Float16)bv[it].y;
                Bs[(n4 * 4 + 2) * LDSS + kk] = (_Float16)bv[it].z;
                Bs[(n4 * 4 + 3) * LDSS + kk] = (_Float16)bv[it].w;
            }
        }
        __syncthreads();

        // ---- stage 3: 8 WMMAs per K-tile; load all 4 frags, then 4 WMMAs ----
        #pragma unroll
        for (int ks = 0; ks < BK; ks += 32) {
            union { v16h v; struct { uint4 lo, hi; } q; } ua0, ua1, ub0, ub1;
            const _Float16* ap0 = &As[(wm * 32 + lr) * LDSS + ks + kb];
            const _Float16* ap1 = ap0 + 16 * LDSS;
            const _Float16* bp0 = &Bs[(wn * 32 + lr) * LDSS + ks + kb];
            const _Float16* bp1 = bp0 + 16 * LDSS;
            ua0.q.lo = *(const uint4*)(ap0); ua0.q.hi = *(const uint4*)(ap0 + 16);
            ua1.q.lo = *(const uint4*)(ap1); ua1.q.hi = *(const uint4*)(ap1 + 16);
            ub0.q.lo = *(const uint4*)(bp0); ub0.q.hi = *(const uint4*)(bp0 + 16);
            ub1.q.lo = *(const uint4*)(bp1); ub1.q.hi = *(const uint4*)(bp1 + 16);
            acc00 = WMMA_F16(ua0.v, ub0.v, acc00);
            acc01 = WMMA_F16(ua0.v, ub1.v, acc01);
            acc10 = WMMA_F16(ua1.v, ub0.v, acc10);
            acc11 = WMMA_F16(ua1.v, ub1.v, acc11);
        }
    }

    // Epilogue: VGPR r -> M = r + 8*half16 ; N = lane%16 (ISA C/D layout)
    const int m0 = rowBase + wm * 32;
    const int n0 = colBase + wn * 32 + lr;
    #pragma unroll
    for (int r = 0; r < 8; ++r) {
        int mA = m0 + r + 8 * half16;       // rows of a0
        int mB = mA + 16;                   // rows of a1
        float v00 = acc00[r], v01 = acc01[r], v10 = acc10[r], v11 = acc11[r];
        if (BIAS) {
            float b0 = bias[n0], b1 = bias[n0 + 16];
            v00 += b0; v01 += b1; v10 += b0; v11 += b1;
        }
        if (GELU_ACT) {
            v00 = 0.5f * v00 * (1.0f + erff(v00 * 0.70710678118654752f));
            v01 = 0.5f * v01 * (1.0f + erff(v01 * 0.70710678118654752f));
            v10 = 0.5f * v10 * (1.0f + erff(v10 * 0.70710678118654752f));
            v11 = 0.5f * v11 * (1.0f + erff(v11 * 0.70710678118654752f));
        }
        if (RES) {
            v00 += Rb[(size_t)mA * N + n0];
            v01 += Rb[(size_t)mA * N + n0 + 16];
            v10 += Rb[(size_t)mB * N + n0];
            v11 += Rb[(size_t)mB * N + n0 + 16];
        }
        Cb[(size_t)mA * N + n0]      = v00;
        Cb[(size_t)mA * N + n0 + 16] = v01;
        Cb[(size_t)mB * N + n0]      = v10;
        Cb[(size_t)mB * N + n0 + 16] = v11;
    }
}

// ------------------------- elementwise / small kernels ---------------------

__global__ void af_mean_time(const float* __restrict__ x, float* __restrict__ mean)
{
    int i = blockIdx.x * blockDim.x + threadIdx.x;
    if (i >= AF_B * AF_CIN) return;
    int b = i / AF_CIN, c = i % AF_CIN;
    float s = 0.0f;
    for (int t = 0; t < AF_L; ++t) s += x[((size_t)b * AF_L + t) * AF_CIN + c];
    mean[i] = s * (1.0f / AF_L);
}

// series_decomp: 25-tap moving mean with replicate (clamp) padding
__global__ void af_decomp(const float* __restrict__ x, float* __restrict__ seasonal,
                          float* __restrict__ trend, int C)
{
    int idx = blockIdx.x * blockDim.x + threadIdx.x;
    int total = AF_B * AF_L * C;
    if (idx >= total) return;
    int c = idx % C;
    int t = (idx / C) % AF_L;
    int b = idx / (C * AF_L);
    float s = 0.0f;
    for (int w = 0; w < AF_MA; ++w) {
        int tt = t + w - (AF_MA - 1) / 2;
        tt = tt < 0 ? 0 : (tt >= AF_L ? AF_L - 1 : tt);
        s += x[((size_t)b * AF_L + tt) * C + c];
    }
    float m = s * (1.0f / AF_MA);
    trend[idx] = m;
    seasonal[idx] = x[idx] - m;
}

__global__ void af_build_inits(const float* __restrict__ seasXE, const float* __restrict__ trendXE,
                               const float* __restrict__ meanE,
                               float* __restrict__ seasInit, float* __restrict__ trendInit)
{
    int idx = blockIdx.x * blockDim.x + threadIdx.x;
    int total = AF_B * AF_L * AF_CIN;
    if (idx >= total) return;
    int c = idx % AF_CIN;
    int t = (idx / AF_CIN) % AF_L;
    int b = idx / (AF_CIN * AF_L);
    if (t < AF_L / 2) {
        size_t src = ((size_t)b * AF_L + (AF_L / 2 + t)) * AF_CIN + c;
        trendInit[idx] = trendXE[src];
        seasInit[idx]  = seasXE[src];
    } else {
        trendInit[idx] = meanE[b * AF_CIN + c];
        seasInit[idx]  = 0.0f;
    }
}

// embedding: circular conv3 token embed + temporal embed
__global__ void af_embed(const float* __restrict__ x, const float* __restrict__ Wtok,
                         const float* __restrict__ mark, const float* __restrict__ Wtem,
                         float* __restrict__ out)
{
    int idx = blockIdx.x * blockDim.x + threadIdx.x;
    int total = AF_B * AF_L * AF_D;
    if (idx >= total) return;
    int d = idx % AF_D;
    int l = (idx / AF_D) % AF_L;
    int b = idx / (AF_D * AF_L);
    float acc = 0.0f;
    for (int j = 0; j < 3; ++j) {
        int tl = (l + j - 1 + AF_L) & (AF_L - 1);
        const float* xr = x + ((size_t)b * AF_L + tl) * AF_CIN;
        const float* wr = Wtok + (size_t)j * AF_CIN * AF_D + d;
        for (int c = 0; c < AF_CIN; ++c) acc += xr[c] * wr[(size_t)c * AF_D];
    }
    const float* mr = mark + ((size_t)b * AF_L + l) * AF_MARK;
    for (int c = 0; c < AF_MARK; ++c) acc += mr[c] * Wtem[(size_t)c * AF_D + d];
    out[idx] = acc;
}

// mean_value[b,tau] = (1/512)*sum_t S[b,(t+tau)%L,t] = (1/512)*sum_r S[b,r,(r-tau)%L]
// (row-major form: adjacent tau lanes read adjacent columns -> coalesced)
__global__ void af_diag_reduce(const float* __restrict__ S, float* __restrict__ mv)
{
    int idx = blockIdx.x * blockDim.x + threadIdx.x;
    if (idx >= AF_B * AF_L) return;
    int b = idx >> 10, tau = idx & (AF_L - 1);
    const float* Sb = S + (size_t)b * AF_L * AF_L;
    float acc = 0.0f;
    for (int r = 0; r < AF_L; ++r)
        acc += Sb[(size_t)r * AF_L + ((r - tau) & (AF_L - 1))];
    mv[idx] = acc * (1.0f / 512.0f);
}

// top-6 + softmax per batch (one lane per batch)
__global__ void af_topk6(const float* __restrict__ mv, float* __restrict__ wts,
                         int* __restrict__ dly)
{
    int b = threadIdx.x;
    if (b >= AF_B) return;
    const float* m = mv + (size_t)b * AF_L;
    float v[AF_TOPK]; int id[AF_TOPK];
    for (int i = 0; i < AF_TOPK; ++i) {
        float best = -3.4e38f; int bi = 0;
        for (int j = 0; j < AF_L; ++j) {
            bool used = false;
            for (int p = 0; p < i; ++p) if (id[p] == j) used = true;
            if (!used && m[j] > best) { best = m[j]; bi = j; }
        }
        v[i] = best; id[i] = bi;
    }
    float mx = v[0], s = 0.0f, e[AF_TOPK];
    for (int i = 0; i < AF_TOPK; ++i) { e[i] = expf(v[i] - mx); s += e[i]; }
    for (int i = 0; i < AF_TOPK; ++i) { wts[b * AF_TOPK + i] = e[i] / s; dly[b * AF_TOPK + i] = id[i]; }
}

// out[b,t,d] = sum_i w[b,i] * V[b,(t+delay_i)%L, d]
__global__ void af_delay_agg(const float* __restrict__ V, const float* __restrict__ wts,
                             const int* __restrict__ dly, float* __restrict__ O)
{
    int idx = blockIdx.x * blockDim.x + threadIdx.x;
    int total = AF_B * AF_L * AF_D;
    if (idx >= total) return;
    int d = idx % AF_D;
    int t = (idx / AF_D) % AF_L;
    int b = idx / (AF_D * AF_L);
    float acc = 0.0f;
    for (int i = 0; i < AF_TOPK; ++i) {
        int tt = (t + dly[b * AF_TOPK + i]) & (AF_L - 1);
        acc += wts[b * AF_TOPK + i] * V[((size_t)b * AF_L + tt) * AF_D + d];
    }
    O[idx] = acc;
}

__global__ void af_ln_rows(const float* __restrict__ X, const float* __restrict__ w,
                           const float* __restrict__ bvec, float* __restrict__ out)
{
    int row = blockIdx.x * blockDim.x + threadIdx.x;
    if (row >= AF_B * AF_L) return;
    const float* x = X + (size_t)row * AF_D;
    float s = 0.0f, ss = 0.0f;
    for (int d = 0; d < AF_D; ++d) { float v = x[d]; s += v; ss += v * v; }
    float mu = s * (1.0f / AF_D);
    float var = ss * (1.0f / AF_D) - mu * mu;
    float inv = 1.0f / sqrtf(var + 1e-5f);
    float* o = out + (size_t)row * AF_D;
    for (int d = 0; d < AF_D; ++d) o[d] = (x[d] - mu) * inv * w[d] + bvec[d];
}

__global__ void af_colmean(const float* __restrict__ X, float* __restrict__ cm)
{
    int i = blockIdx.x * blockDim.x + threadIdx.x;
    if (i >= AF_B * AF_D) return;
    int b = i / AF_D, d = i % AF_D;
    float s = 0.0f;
    for (int t = 0; t < AF_L; ++t) s += X[((size_t)b * AF_L + t) * AF_D + d];
    cm[i] = s * (1.0f / AF_L);
}

__global__ void af_sub_colmean(const float* __restrict__ X, const float* __restrict__ cm,
                               float* __restrict__ out)
{
    int idx = blockIdx.x * blockDim.x + threadIdx.x;
    int total = AF_B * AF_L * AF_D;
    if (idx >= total) return;
    int d = idx % AF_D;
    int b = idx / (AF_D * AF_L);
    out[idx] = X[idx] - cm[b * AF_D + d];
}

__global__ void af_add3(const float* __restrict__ a, const float* __restrict__ b,
                        const float* __restrict__ c, float* __restrict__ o, int n)
{
    int i = blockIdx.x * blockDim.x + threadIdx.x;
    if (i < n) o[i] = a[i] + b[i] + c[i];
}

// trend_out = trend_in + circ_conv3(tsum[512ch], Wt[3,512,7])
__global__ void af_trend_conv_add(const float* __restrict__ ts, const float* __restrict__ Wt,
                                  const float* __restrict__ trIn, float* __restrict__ trOut)
{
    int idx = blockIdx.x * blockDim.x + threadIdx.x;
    int total = AF_B * AF_L * AF_CIN;
    if (idx >= total) return;
    int c = idx % AF_CIN;
    int t = (idx / AF_CIN) % AF_L;
    int b = idx / (AF_CIN * AF_L);
    float acc = trIn[idx];
    for (int j = 0; j < 3; ++j) {
        int tt = (t + j - 1 + AF_L) & (AF_L - 1);
        const float* xr = ts + ((size_t)b * AF_L + tt) * AF_D;
        const float* wr = Wt + (size_t)j * AF_D * AF_CIN + c;
        for (int d = 0; d < AF_D; ++d) acc += xr[d] * wr[(size_t)d * AF_CIN];
    }
    trOut[idx] = acc;
}

// out[b,t',c] = trend[b,512+t',c] + decLN[b,512+t',:] @ projW + projB
__global__ void af_final(const float* __restrict__ decLN, const float* __restrict__ pW,
                         const float* __restrict__ pb, const float* __restrict__ trend,
                         float* __restrict__ out)
{
    int idx = blockIdx.x * blockDim.x + threadIdx.x;
    int total = AF_B * (AF_L / 2) * AF_CIN;
    if (idx >= total) return;
    int c = idx % AF_CIN;
    int tp = (idx / AF_CIN) % (AF_L / 2);
    int b = idx / (AF_CIN * (AF_L / 2));
    int t = AF_L / 2 + tp;
    float acc = pb[c] + trend[((size_t)b * AF_L + t) * AF_CIN + c];
    const float* xr = decLN + ((size_t)b * AF_L + t) * AF_D;
    for (int d = 0; d < AF_D; ++d) acc += xr[d] * pW[(size_t)d * AF_CIN + c];
    out[idx] = acc;
}

// ----------------------------- host orchestration --------------------------

struct AttnWs { float *Q, *K, *V, *S, *mv, *wts, *O; int* dly; };

static void run_attention(const float* xq, const float* xkv,
                          const float* W, const float* bvec,
                          float* outBuf, const AttnWs& w, hipStream_t s)
{
    const int M = AF_B * AF_L, D = AF_D;
    dim3 blk(256);
    dim3 gP(D / BN, M / BM, 1);
    gemm_wmma_f16<false,true,false,false><<<gP, blk, 0, s>>>(xq,  W,             bvec,         nullptr, w.Q, M, D, D, 0, 0, 0);
    gemm_wmma_f16<false,true,false,false><<<gP, blk, 0, s>>>(xkv, W + D * D,     bvec + D,     nullptr, w.K, M, D, D, 0, 0, 0);
    gemm_wmma_f16<false,true,false,false><<<gP, blk, 0, s>>>(xkv, W + 2 * D * D, bvec + 2 * D, nullptr, w.V, M, D, D, 0, 0, 0);
    // batched S[b] = Q[b] * K[b]^T  (M=N=1024, K=512)
    dim3 gS(AF_L / BN, AF_L / BM, AF_B);
    gemm_wmma_f16<true,false,false,false><<<gS, blk, 0, s>>>(
        w.Q, w.K, nullptr, nullptr, w.S, AF_L, AF_L, AF_D,
        (long long)AF_L * AF_D, (long long)AF_L * AF_D, (long long)AF_L * AF_L);
    af_diag_reduce<<<(AF_B * AF_L) / 256, 256, 0, s>>>(w.S, w.mv);
    af_topk6<<<1, 32, 0, s>>>(w.mv, w.wts, w.dly);
    af_delay_agg<<<(AF_B * AF_L * AF_D) / 256, 256, 0, s>>>(w.V, w.wts, w.dly, w.O);
    // O-proj + bias + residual(xq)
    gemm_wmma_f16<false,true,false,true><<<gP, blk, 0, s>>>(w.O, W + 3 * D * D, bvec + 3 * D, xq, outBuf, M, D, D, 0, 0, 0);
}

extern "C" void kernel_launch(void* const* d_in, const int* in_sizes, int n_in,
                              void* d_out, int out_size, void* d_ws, size_t ws_size,
                              hipStream_t stream)
{
    (void)in_sizes; (void)n_in; (void)out_size; (void)ws_size;

    const float* x_enc       = (const float*)d_in[0];
    const float* x_mark_enc  = (const float*)d_in[1];
    /* d_in[2] = x_dec (unused numerically) */
    const float* x_mark_dec  = (const float*)d_in[3];
    const float* W_enc_tok   = (const float*)d_in[4];
    const float* W_enc_tem   = (const float*)d_in[5];
    const float* W_dec_tok   = (const float*)d_in[6];
    const float* W_dec_tem   = (const float*)d_in[7];
    const float* enc_attn_W  = (const float*)d_in[8];
    const float* enc_attn_b  = (const float*)d_in[9];
    const float* enc_ff1     = (const float*)d_in[10];
    const float* enc_ff2     = (const float*)d_in[11];
    const float* enc_ln_w    = (const float*)d_in[12];
    const float* enc_ln_b    = (const float*)d_in[13];
    const float* dec_self_W  = (const float*)d_in[14];
    const float* dec_self_b  = (const float*)d_in[15];
    const float* dec_cross_W = (const float*)d_in[16];
    const float* dec_cross_b = (const float*)d_in[17];
    const float* dec_ff1     = (const float*)d_in[18];
    const float* dec_ff2     = (const float*)d_in[19];
    const float* dec_trend_W = (const float*)d_in[20];
    const float* dec_ln_w    = (const float*)d_in[21];
    const float* dec_ln_b    = (const float*)d_in[22];
    const float* proj_W      = (const float*)d_in[23];
    const float* proj_b      = (const float*)d_in[24];
    float* out = (float*)d_out;

    const size_t SZ_BLD = (size_t)AF_B * AF_L * AF_D;
    const size_t SZ_MID = (size_t)AF_B * AF_L * AF_FF;
    const size_t SZ_S   = (size_t)AF_B * AF_L * AF_L;
    const size_t SZ_SM  = (size_t)AF_B * AF_L * AF_CIN;

    char* p = (char*)d_ws;
    auto alloc = [&](size_t bytes) -> void* {
        void* r = (void*)p;
        p += (bytes + 255) & ~(size_t)255;
        return r;
    };

    float* bufA   = (float*)alloc(SZ_BLD * 4);
    float* bufB   = (float*)alloc(SZ_BLD * 4);
    float* bufC   = (float*)alloc(SZ_BLD * 4);
    float* ENCF   = (float*)alloc(SZ_BLD * 4);
    float* Q      = (float*)alloc(SZ_BLD * 4);
    float* K      = (float*)alloc(SZ_BLD * 4);
    float* V      = (float*)alloc(SZ_BLD * 4);
    float* O      = (float*)alloc(SZ_BLD * 4);
    float* T1     = (float*)alloc(SZ_BLD * 4);
    float* T2     = (float*)alloc(SZ_BLD * 4);
    float* T3     = (float*)alloc(SZ_BLD * 4);
    float* MID    = (float*)alloc(SZ_MID * 4);
    float* S      = (float*)alloc(SZ_S * 4);
    float* mv     = (float*)alloc((size_t)AF_B * AF_L * 4);
    float* wts    = (float*)alloc((size_t)AF_B * AF_TOPK * 4);
    int*   dly    = (int*)  alloc((size_t)AF_B * AF_TOPK * 4);
    float* meanE  = (float*)alloc((size_t)AF_B * AF_CIN * 4);
    float* seasXE = (float*)alloc(SZ_SM * 4);
    float* trendXE= (float*)alloc(SZ_SM * 4);
    float* seasI  = (float*)alloc(SZ_SM * 4);
    float* trendI = (float*)alloc(SZ_SM * 4);
    float* trendA = (float*)alloc(SZ_SM * 4);
    float* colm   = (float*)alloc((size_t)AF_B * AF_D * 4);

    AttnWs aw{Q, K, V, S, mv, wts, O, dly};

    const int M = AF_B * AF_L;
    dim3 blk(256);
    dim3 gD(AF_D / BN, M / BM, 1);    // N=512 GEMMs
    dim3 gF(AF_FF / BN, M / BM, 1);   // N=2048 GEMMs
    int nBLD = (int)(SZ_BLD / 256);
    int nSM  = (int)((SZ_SM + 255) / 256);

    // ----- input decomposition and init series -----
    af_mean_time<<<1, 256, 0, stream>>>(x_enc, meanE);
    af_decomp<<<nSM, 256, 0, stream>>>(x_enc, seasXE, trendXE, AF_CIN);
    af_build_inits<<<nSM, 256, 0, stream>>>(seasXE, trendXE, meanE, seasI, trendI);

    // ----- encoder -----
    af_embed<<<nBLD, 256, 0, stream>>>(x_enc, W_enc_tok, x_mark_enc, W_enc_tem, bufA);
    for (int l = 0; l < 2; ++l) {
        const float* Wl = enc_attn_W + (size_t)l * 4 * AF_D * AF_D;
        const float* bl = enc_attn_b + (size_t)l * 4 * AF_D;
        run_attention(bufA, bufA, Wl, bl, bufB, aw, stream);
        af_decomp<<<nBLD, 256, 0, stream>>>(bufB, bufA, T1, AF_D);
        gemm_wmma_f16<false,false,true,false><<<gF, blk, 0, stream>>>(
            bufA, enc_ff1 + (size_t)l * AF_D * AF_FF, nullptr, nullptr,
            MID, M, AF_FF, AF_D, 0, 0, 0);
        gemm_wmma_f16<false,false,false,true><<<gD, blk, 0, stream>>>(
            MID, enc_ff2 + (size_t)l * AF_FF * AF_D, nullptr, bufA,
            bufB, M, AF_D, AF_FF, 0, 0, 0);
        af_decomp<<<nBLD, 256, 0, stream>>>(bufB, bufA, T1, AF_D);
    }
    af_ln_rows<<<M / 256, 256, 0, stream>>>(bufA, enc_ln_w, enc_ln_b, bufB);
    af_colmean<<<(AF_B * AF_D) / 256, 256, 0, stream>>>(bufB, colm);
    af_sub_colmean<<<nBLD, 256, 0, stream>>>(bufB, colm, ENCF);

    // ----- decoder -----
    af_embed<<<nBLD, 256, 0, stream>>>(seasI, W_dec_tok, x_mark_dec, W_dec_tem, bufA);
    // self attention
    run_attention(bufA, bufA, dec_self_W, dec_self_b, bufB, aw, stream);
    af_decomp<<<nBLD, 256, 0, stream>>>(bufB, bufA, T1, AF_D);
    // cross attention against encoder output
    run_attention(bufA, ENCF, dec_cross_W, dec_cross_b, bufB, aw, stream);
    af_decomp<<<nBLD, 256, 0, stream>>>(bufB, bufA, T2, AF_D);
    // feed-forward
    gemm_wmma_f16<false,false,true,false><<<gF, blk, 0, stream>>>(
        bufA, dec_ff1, nullptr, nullptr, MID, M, AF_FF, AF_D, 0, 0, 0);
    gemm_wmma_f16<false,false,false,true><<<gD, blk, 0, stream>>>(
        MID, dec_ff2, nullptr, bufA, bufB, M, AF_D, AF_FF, 0, 0, 0);
    af_decomp<<<nBLD, 256, 0, stream>>>(bufB, bufA, T3, AF_D);
    // trend path
    af_add3<<<nBLD, 256, 0, stream>>>(T1, T2, T3, bufC, (int)SZ_BLD);
    af_trend_conv_add<<<nSM, 256, 0, stream>>>(bufC, dec_trend_W, trendI, trendA);
    // final layernorm + projection + slice
    af_ln_rows<<<M / 256, 256, 0, stream>>>(bufA, dec_ln_w, dec_ln_b, bufB);
    af_colmean<<<(AF_B * AF_D) / 256, 256, 0, stream>>>(bufB, colm);
    af_sub_colmean<<<nBLD, 256, 0, stream>>>(bufB, colm, bufC);
    int nFin = (AF_B * (AF_L / 2) * AF_CIN + 255) / 256;
    af_final<<<nFin, 256, 0, stream>>>(bufC, proj_W, proj_b, trendA, out);
}
